// VanillaVectorQuantizer_58411555225657
// MI455X (gfx1250) — matrix operand order repaired and checked by reading, not compiled, not run on hardware
//
#include <hip/hip_runtime.h>

typedef __attribute__((ext_vector_type(16))) _Float16     v16h;
typedef __attribute__((ext_vector_type(2)))  __fp16       v2fp16;
typedef __attribute__((ext_vector_type(8)))  float        v8f;
typedef __attribute__((ext_vector_type(4)))  unsigned int v4u;
typedef __attribute__((ext_vector_type(8)))  int          v8i;
typedef __attribute__((ext_vector_type(4)))  int          v4i;

#define D_DIM   64
#define K_CODES 512
#define HW      4096
#define COLS    64      // spatial columns per block
#define THREADS 128     // 4 waves of 32

// ---------------------------------------------------------------------------
// Prep: codebook [D=64][K=512] f32 -> code-major f16 [K][D] in ws, plus
//       nhsq[k] = -0.5*||e_k||^2 (negated so it can seed the WMMA C operand).
// ---------------------------------------------------------------------------
__global__ __launch_bounds__(K_CODES)
void vq_prep(const float* __restrict__ cb,
             _Float16* __restrict__ cb16,
             float* __restrict__ nhsq)
{
    const int k = threadIdx.x;          // 0..511
    float acc = 0.0f;
#pragma unroll 4
    for (int d = 0; d < D_DIM; ++d) {
        float v = cb[d * K_CODES + k];
        acc += v * v;
        cb16[k * D_DIM + d] = (_Float16)v;
    }
    nhsq[k] = -0.5f * acc;
}

// ---------------------------------------------------------------------------
// Main: per block, one batch's tile of 64 spatial columns.
//   score[k][n] = -0.5||e_k||^2 + sum_d e16[k][d]*x16[d][n]
//   (C operand seeded with -0.5||e||^2; argmax == argmin of the distance)
// ---------------------------------------------------------------------------
__global__ __launch_bounds__(THREADS)
void vq_main(const float* __restrict__ x,        // [B,64,4096]
             const float* __restrict__ cb,       // [64,512] f32 (for gather)
             const _Float16* __restrict__ cb16,  // [512,64] f16 (A operand)
             const float* __restrict__ nhsq,     // [512] = -0.5||e||^2
             float* __restrict__ out)            // [B,64,4096]
{
    __shared__ float Xs[D_DIM * COLS];     // f32 X tile (TDM destination)
    __shared__ float Hs[K_CODES];          // -0.5||e||^2
    __shared__ int   Ids[COLS];            // winning code per column

    const int tid   = threadIdx.x;
    const int blk   = blockIdx.x;
    const int batch = blk / (HW / COLS);
    const int col0  = (blk % (HW / COLS)) * COLS;
    const size_t base = (size_t)batch * D_DIM * HW + col0;

    // Warm the f16 codebook (64KB) into WGP$/L2: global_prefetch_b8 path.
    __builtin_prefetch(cb16 + (tid << 8), 0, 0);

#if __has_builtin(__builtin_amdgcn_tensor_load_to_lds) && \
    __has_builtin(__builtin_amdgcn_s_wait_tensorcnt)
    // Tensor Data Mover: DMA the 64(d) x 64(col) f32 tile (row stride 4096)
    // from global into LDS. Descriptor per ISA 8.3/8.4; wave 0 issues it.
    if (tid < 32) {
        unsigned long long ga =
            (unsigned long long)(uintptr_t)(const void*)(x + base);
        unsigned int ldsOff = (unsigned int)(uintptr_t)
            (__attribute__((address_space(3))) float*)Xs;
        // Group 0: count=1 | lds_addr | global_addr[56:0] | type=2 (bits 127:126)
        v4u g0 = { 1u,
                   ldsOff,
                   (unsigned int)ga,
                   (unsigned int)((ga >> 32) & 0x01FFFFFFu) | (2u << 30) };
        // Group 1: wg_mask=0, data_size=2 (4B); tensor_dim0=4096, tensor_dim1=64;
        // tile_dim0=64 (cols), tile_dim1=64 (rows); tensor_dim0_stride=4096.
        v8i g1 = { (int)0x20000u,                    // data_size=4B, flags 0
                   (int)((HW & 0xFFFF) << 16),       // tensor_dim0 lo16 @ bits63:48
                   (int)(D_DIM << 16),               // tensor_dim1 lo16 @ bits95:80
                   (int)(COLS << 16),                // tile_dim0     @ bits127:112
                   (int)D_DIM,                       // tile_dim1     @ bits143:128
                   (int)HW,                          // tensor_dim0_stride lo32
                   0, 0 };
        v4i gz4 = { 0, 0, 0, 0 };                    // groups 2/3: dims unused
        v8i gz8 = { 0, 0, 0, 0, 0, 0, 0, 0 };        // trailing group (zero-fill)
        __builtin_amdgcn_tensor_load_to_lds(g0, g1, gz4, gz4, gz8, 0);
        __builtin_amdgcn_s_wait_tensorcnt(0);        // s_wait_tensorcnt 0
    }
#else
    for (int i = tid; i < D_DIM * COLS; i += THREADS) {
        int d = i >> 6, c = i & (COLS - 1);
        Xs[i] = x[base + (size_t)d * HW + c];
    }
#endif
    for (int i = tid; i < K_CODES; i += THREADS) Hs[i] = nhsq[i];
    __syncthreads();

    const int lane  = tid & 31;
    const int wave  = tid >> 5;
    const int half  = lane >> 4;          // which 16-lane half
    const int l15   = lane & 15;
    const int mycol = (wave << 4) + l15;  // column owned by this lane (per half)

    // B operand (32x16, K=d x N=col), contiguous k order: VGPR v holds K=2v,2v+1 for
    // lanes 0-15 and K=16+2v,16+2v+1 for lanes 16-31; N = lane&15.
    union F16x16 { v16h v; v2fp16 p[8]; _Float16 h[16]; };
    F16x16 bfrag[2];                      // g=0: d 0..31, g=1: d 32..63
#pragma unroll
    for (int g = 0; g < 2; ++g) {
        const int dbase = g * 32 + half * 16;
#pragma unroll
        for (int vv = 0; vv < 8; ++vv) {
            float f0 = Xs[(dbase + 2 * vv)     * COLS + mycol];
            float f1 = Xs[(dbase + 2 * vv + 1) * COLS + mycol];
#if __has_builtin(__builtin_amdgcn_cvt_pkrtz)
            bfrag[g].p[vv] = __builtin_amdgcn_cvt_pkrtz(f0, f1);
#else
            bfrag[g].h[2 * vv]     = (_Float16)f0;
            bfrag[g].h[2 * vv + 1] = (_Float16)f1;
#endif
        }
    }

    float best   = -3.0e38f;
    int   bestid = 0;

    // 32 code tiles of 16; A fragment = four 16B global loads per lane
    // (K 0-15 for lanes 0-15, K 16-31 for lanes 16-31, contiguous).
    for (int m = 0; m < K_CODES / 16; ++m) {
        const _Float16* arow = cb16 + ((m << 4) + l15) * D_DIM;
        const int mrow = (m << 4) + (half << 3);

        // Seed C with -0.5||e||^2 for the 8 codes this lane holds
        // (C layout: VGPR r -> M=r for lanes 0-15, M=r+8 for lanes 16-31).
        union { uint4 q[2]; v8f v; } cinit;
        cinit.q[0] = *(const uint4*)(Hs + mrow);
        cinit.q[1] = *(const uint4*)(Hs + mrow + 4);
        v8f c = cinit.v;

#pragma unroll
        for (int g = 0; g < 2; ++g) {
            union { uint4 q[2]; v16h v; } a;
            a.q[0] = *(const uint4*)(arow + g * 32 + half * 16);
            a.q[1] = *(const uint4*)(arow + g * 32 + half * 16 + 8);
            c = __builtin_amdgcn_wmma_f32_16x16x32_f16(
                    false, a.v, false, bfrag[g].v, (short)0, c, false, false);
        }
#pragma unroll
        for (int r = 0; r < 8; ++r) {
            if (c[r] > best) { best = c[r]; bestid = mrow + r; }
        }
    }

    // Each column's candidates live in lane pair (l, l^16): reduce.
    float ob = __shfl_xor(best, 16, 32);
    int   oi = __shfl_xor(bestid, 16, 32);
    if (ob > best || (ob == best && oi < bestid)) { best = ob; bestid = oi; }
    if (half == 0) Ids[mycol] = bestid;
    __syncthreads();

    // Gather f32 codebook rows -> output (writes coalesced along columns;
    // cb reads are scattered within a 2KB row, L2-resident).
    for (int i = tid; i < D_DIM * COLS; i += THREADS) {
        int d = i >> 6, c = i & (COLS - 1);
        out[base + (size_t)d * HW + c] = cb[d * K_CODES + Ids[c]];
    }
}

// ---------------------------------------------------------------------------
extern "C" void kernel_launch(void* const* d_in, const int* in_sizes, int n_in,
                              void* d_out, int out_size, void* d_ws, size_t ws_size,
                              hipStream_t stream)
{
    const float* x  = (const float*)d_in[0];   // encodings [32,64,64,64]
    const float* cb = (const float*)d_in[1];   // codebook  [64,512]

    _Float16* cb16 = (_Float16*)d_ws;                                   // 64 KB
    float*    nhsq = (float*)((char*)d_ws + K_CODES * D_DIM * sizeof(_Float16));

    vq_prep<<<1, K_CODES, 0, stream>>>(cb, cb16, nhsq);

    const int blocks = in_sizes[0] / (D_DIM * COLS);   // 2048 blocks of 64 cols
    vq_main<<<blocks, THREADS, 0, stream>>>(x, cb, cb16, nhsq, (float*)d_out);
}
